// EnhancedSegmentEncoder_39685497815365
// MI455X (gfx1250) — matrix operand order repaired and checked by reading, not compiled, not run on hardware
//
#include <hip/hip_runtime.h>
#include <hip/hip_bf16.h>
#include <math.h>

// Problem constants (from the reference)
#define SQ   512          // sequence length S
#define BQ   4            // batch B
#define DINC 768          // input dim
#define DM   1024         // model dim D
#define NH   16           // heads
#define HDIM 64           // head dim
#define FF   4096         // ffn dim
#define NL   12           // layers
#define MROWS (SQ*BQ)     // 2048 token rows

typedef __bf16 bf16_t;
typedef __attribute__((ext_vector_type(16))) __bf16 v16bf;
typedef __attribute__((ext_vector_type(8)))  __bf16 v8bf;
typedef __attribute__((ext_vector_type(8)))  float  v8f;

// ---------------------------------------------------------------------------
// f32 -> bf16 conversion (vectorized by 4; all our sizes are multiples of 4)
// ---------------------------------------------------------------------------
__global__ void cvt_kernel(const float* __restrict__ src, bf16_t* __restrict__ dst,
                           long long n) {
  long long i = ((long long)blockIdx.x * blockDim.x + threadIdx.x) * 4;
  long long stride = (long long)gridDim.x * blockDim.x * 4;
  for (; i < n; i += stride) {
    float4 f = *(const float4*)(src + i);
    dst[i+0] = (bf16_t)f.x;
    dst[i+1] = (bf16_t)f.y;
    dst[i+2] = (bf16_t)f.z;
    dst[i+3] = (bf16_t)f.w;
  }
}

// ---------------------------------------------------------------------------
// LayerNorm over D=1024, one block per token row, bf16 output for WMMA input
// ---------------------------------------------------------------------------
__global__ __launch_bounds__(256) void ln_kernel(const float* __restrict__ x,
                                                 const float* __restrict__ g,
                                                 const float* __restrict__ b,
                                                 bf16_t* __restrict__ out) {
  const int row = blockIdx.x;
  const int t = threadIdx.x;
  const float* xr = x + (long long)row * DM;
  float vals[4];
  float s = 0.f, sq = 0.f;
#pragma unroll
  for (int j = 0; j < 4; ++j) {
    float v = xr[t + j*256];
    vals[j] = v; s += v; sq += v*v;
  }
  __shared__ float rs[256], rq[256];
  rs[t] = s; rq[t] = sq; __syncthreads();
  for (int off = 128; off > 0; off >>= 1) {
    if (t < off) { rs[t] += rs[t+off]; rq[t] += rq[t+off]; }
    __syncthreads();
  }
  const float mean = rs[0] * (1.0f/DM);
  const float var  = rq[0] * (1.0f/DM) - mean*mean;
  const float inv  = rsqrtf(var + 1e-5f);
#pragma unroll
  for (int j = 0; j < 4; ++j) {
    int d = t + j*256;
    out[(long long)row*DM + d] = (bf16_t)((vals[j]-mean)*inv*g[d] + b[d]);
  }
}

// ---------------------------------------------------------------------------
// x += durations*Wdur + bdur (rank-1 embed term; bproj handled in GEMM bias)
// ---------------------------------------------------------------------------
__global__ void embed_extras_kernel(float* __restrict__ x,
                                    const float* __restrict__ dur,
                                    const float* __restrict__ Wdur,
                                    const float* __restrict__ bdur) {
  int idx = blockIdx.x * blockDim.x + threadIdx.x;
  if (idx >= MROWS*DM) return;
  int d = idx % DM;
  int m = idx / DM;
  x[idx] += dur[m] * Wdur[d] + bdur[d];
}

// ---------------------------------------------------------------------------
// RoPE (rotate=1) or plain transpose (rotate=0):
//   in : f32 (s,b, h*HD+hd)  ->  out : bf16 [b][h][s][hd]
// ---------------------------------------------------------------------------
__global__ void rope_kernel(const float* __restrict__ in, bf16_t* __restrict__ out,
                            int rotate) {
  int idx = blockIdx.x * blockDim.x + threadIdx.x;
  if (idx >= MROWS*DM) return;
  int hd = idx % HDIM;
  int hh = (idx / HDIM) % NH;
  int b  = (idx / DM) % BQ;
  int s  = idx / (BQ*DM);
  float v = in[idx];
  if (rotate) {
    int f = hd & 31;
    float invf = powf(10000.0f, -(float)f * (1.0f/32.0f));
    float ang  = (float)s * invf;
    float c = cosf(ang), sn = sinf(ang);
    float other = (hd < 32) ? -in[idx + 32] : in[idx - 32];
    v = v * c + other * sn;
  }
  out[((((long long)b*NH + hh)*SQ + s)*HDIM) + hd] = (bf16_t)v;
}

// ---------------------------------------------------------------------------
// Masked scaled softmax over the key axis (S=512), bf16 probs output.
// grid = (S, B*H), block = 256 (2 keys per thread)
// ---------------------------------------------------------------------------
__global__ __launch_bounds__(256) void softmax_kernel(
    const float* __restrict__ scores, const unsigned char* __restrict__ mask,
    bf16_t* __restrict__ probs, float scale) {
  const int q = blockIdx.x;
  const int z = blockIdx.y;          // b*H + h
  const int b = z / NH;
  const long long base = ((long long)z * SQ + q) * SQ;
  const int t = threadIdx.x;
  const int k0 = t, k1 = t + 256;
  float v0 = mask[b*SQ + k0] ? -__builtin_inff() : scores[base + k0] * scale;
  float v1 = mask[b*SQ + k1] ? -__builtin_inff() : scores[base + k1] * scale;
  __shared__ float red[256];
  red[t] = fmaxf(v0, v1); __syncthreads();
  for (int off = 128; off > 0; off >>= 1) {
    if (t < off) red[t] = fmaxf(red[t], red[t+off]);
    __syncthreads();
  }
  const float mx = red[0]; __syncthreads();
  float e0 = __expf(v0 - mx), e1 = __expf(v1 - mx);
  red[t] = e0 + e1; __syncthreads();
  for (int off = 128; off > 0; off >>= 1) {
    if (t < off) red[t] += red[t+off];
    __syncthreads();
  }
  const float inv = 1.0f / red[0];
  probs[base + k0] = (bf16_t)(e0 * inv);
  probs[base + k1] = (bf16_t)(e1 * inv);
}

// ---------------------------------------------------------------------------
// Generic batched bf16 WMMA GEMM, double-buffered LDS pipeline.
//   C[z] = A[z] (MxK) * B[z] (KxN)  (+ bias / gelu / residual epilogue)
// Tiling is templated: WAVES_M x WAVES_N waves, each owning a
// (WM_T*16) x (WN_T*16) tile of WMMA accumulators; block tile
// BM = WAVES_M*WM_T*16, BN = WAVES_N*WN_T*16; K-step 32.
// B is held transposed in LDS ([n][k]) so both A and B fragments are built
// with two ds_read_b128 per 16x16 operand, matching the documented CDNA5
// bf16 WMMA VGPR layouts:
//   A frag element i -> K = (i<8 ? i : i+8) + 8h   (h = lane>>4)
//   B frag element i -> K = (i<8 ? i : i+8) + 8h
//   C frag element e -> row = e + 8h, col = lane&15
// One __syncthreads per K-step: global loads for tile n+1 are issued into
// registers before the WMMAs of tile n, then stored to the other LDS buffer.
// C offset per z: (z/cZdiv)*cOuter + (z%cZdiv)*cInner (lets us scatter the
// per-head attention output straight back into the (s,b,d) layout).
// ---------------------------------------------------------------------------
template <int WAVES_M, int WAVES_N, int WM_T, int WN_T>
__global__ __launch_bounds__(WAVES_M*WAVES_N*32) void wmma_gemm_kernel(
    const bf16_t* __restrict__ A, int lda, long long aBS,
    const bf16_t* __restrict__ Bm, int ldb, long long bBS, int bTrans,
    const float* __restrict__ bias,   // per-column, nullable
    const float* __restrict__ resid,  // same layout as outF, nullable
    float*  __restrict__ outF,        // nullable
    bf16_t* __restrict__ outB,        // nullable
    int ldc, int cZdiv, long long cOuter, long long cInner,
    int K, int doGelu)
{
  constexpr int T     = WAVES_M*WAVES_N*32;
  constexpr int BM    = WAVES_M*WM_T*16;
  constexpr int BN    = WAVES_N*WN_T*16;
  constexpr int EPT_A = BM*32 / T;    // bf16 elems of A tile per thread (16/32)
  constexpr int EPT_B = BN*32 / T;    // bf16 elems of B tile per thread
  constexpr int TPR_B = BN / EPT_B;   // threads per K-row (transpose path)

  __shared__ __align__(16) bf16_t As[2][BM*32];
  __shared__ __align__(16) bf16_t Bt[2][BN*32];   // [n][k]

  const int tid  = threadIdx.x;
  const int wave = tid >> 5;
  const int lane = tid & 31;
  const int h    = lane >> 4;
  const int r    = lane & 15;
  const int wm   = (wave / WAVES_N) * WM_T * 16;
  const int wn   = (wave % WAVES_N) * WN_T * 16;

  const int z      = blockIdx.z;
  const int blockM = blockIdx.y * BM;
  const int blockN = blockIdx.x * BN;

  const bf16_t* Az = A  + (long long)z * aBS;
  const bf16_t* Bz = Bm + (long long)z * bBS;
  const long long cOff = (long long)(z / cZdiv) * cOuter
                       + (long long)(z % cZdiv) * cInner;

  v8f acc[WM_T][WN_T] = {};

  v8bf aReg[EPT_A/8];
  v8bf bReg[EPT_B/8];

  // ---- staged loads: global -> regs, regs -> LDS --------------------------
  auto loadRegs = [&](int k0) {
#pragma unroll
    for (int c = 0; c < EPT_A/16; ++c) {
      const int idx = tid*EPT_A + c*16;
      const int row = idx >> 5, col = idx & 31;
      const bf16_t* src = Az + (long long)(blockM + row)*lda + (k0 + col);
      aReg[c*2]   = *(const v8bf*)(src);
      aReg[c*2+1] = *(const v8bf*)(src + 8);
      if (k0 + 32 < K) __builtin_prefetch(src + 32, 0, 3); // global_prefetch_b8
    }
    if (bTrans) {   // B given as N x K row-major (e.g. K-matrix for scores)
#pragma unroll
      for (int c = 0; c < EPT_B/16; ++c) {
        const int idx = tid*EPT_B + c*16;
        const int row = idx >> 5, col = idx & 31;
        const bf16_t* src = Bz + (long long)(blockN + row)*ldb + (k0 + col);
        bReg[c*2]   = *(const v8bf*)(src);
        bReg[c*2+1] = *(const v8bf*)(src + 8);
      }
    } else {        // B given as K x N row-major: registers hold an n-run
      const int k  = tid / TPR_B;            // 0..31
      const int n0 = (tid % TPR_B) * EPT_B;
      const bf16_t* src = Bz + (long long)(k0 + k)*ldb + (blockN + n0);
#pragma unroll
      for (int c = 0; c < EPT_B/8; ++c)
        bReg[c] = *(const v8bf*)(src + c*8);
      if (k0 + 32 < K) __builtin_prefetch(src + (long long)32*ldb, 0, 3);
    }
  };

  auto storeLds = [&](int buf) {
#pragma unroll
    for (int c = 0; c < EPT_A/16; ++c) {
      const int idx = tid*EPT_A + c*16;
      *(v8bf*)&As[buf][idx]     = aReg[c*2];
      *(v8bf*)&As[buf][idx + 8] = aReg[c*2+1];
    }
    if (bTrans) {
#pragma unroll
      for (int c = 0; c < EPT_B/16; ++c) {
        const int idx = tid*EPT_B + c*16;
        *(v8bf*)&Bt[buf][idx]     = bReg[c*2];
        *(v8bf*)&Bt[buf][idx + 8] = bReg[c*2+1];
      }
    } else {        // transpose scatter into Bt[n][k]
      const int k  = tid / TPR_B;
      const int n0 = (tid % TPR_B) * EPT_B;
#pragma unroll
      for (int c = 0; c < EPT_B/8; ++c)
#pragma unroll
        for (int i = 0; i < 8; ++i)
          Bt[buf][(n0 + c*8 + i)*32 + k] = bReg[c][i];
    }
  };

  // ---- pipelined main loop ------------------------------------------------
  loadRegs(0);
  storeLds(0);
  __syncthreads();

  int buf = 0;
  for (int k0 = 0; k0 < K; k0 += 32) {
    const bool hasNext = (k0 + 32 < K);
    if (hasNext) loadRegs(k0 + 32);

    v16bf afrag[WM_T], bfrag[WN_T];
#pragma unroll
    for (int i = 0; i < WM_T; ++i) {
      const bf16_t* ab = &As[buf][(wm + i*16 + r) * 32];
      v8bf alo = *(const v8bf*)(ab + 8*h);
      v8bf ahi = *(const v8bf*)(ab + 16 + 8*h);
      afrag[i] = __builtin_shufflevector(alo, ahi,
                   0,1,2,3,4,5,6,7,8,9,10,11,12,13,14,15);
    }
#pragma unroll
    for (int j = 0; j < WN_T; ++j) {
      const bf16_t* bb = &Bt[buf][(wn + j*16 + r) * 32];
      v8bf blo = *(const v8bf*)(bb + 8*h);
      v8bf bhi = *(const v8bf*)(bb + 16 + 8*h);
      bfrag[j] = __builtin_shufflevector(blo, bhi,
                   0,1,2,3,4,5,6,7,8,9,10,11,12,13,14,15);
    }
#pragma unroll
    for (int i = 0; i < WM_T; ++i)
#pragma unroll
      for (int j = 0; j < WN_T; ++j)
        acc[i][j] = __builtin_amdgcn_wmma_f32_16x16x32_bf16(
            false, afrag[i], false, bfrag[j],
            (short)0, acc[i][j], false, false);

    if (hasNext) storeLds(buf ^ 1);
    __syncthreads();
    buf ^= 1;
  }

  // ---- epilogue -----------------------------------------------------------
#pragma unroll
  for (int i = 0; i < WM_T; ++i) {
#pragma unroll
    for (int j = 0; j < WN_T; ++j) {
      const int col = blockN + wn + j*16 + r;
#pragma unroll
      for (int e = 0; e < 8; ++e) {
        const int row = blockM + wm + i*16 + e + 8*h;
        float v = acc[i][j][e];
        if (bias)   v += bias[col];
        if (doGelu) v = 0.5f * v * (1.0f + erff(v * 0.70710678118654752f));
        const long long idx = cOff + (long long)row * ldc + col;
        if (resid) v += resid[idx];
        if (outF)  outF[idx] = v;
        if (outB)  outB[idx] = (bf16_t)v;
      }
    }
  }
}

// ---------------------------------------------------------------------------
// Host orchestration
// ---------------------------------------------------------------------------
// Big tiles: 8 waves, wave tile 64x32, block tile 128x128 (256 threads)
// Att tiles: 4 waves, wave tile 32x64, block tile 128x64  (128 threads)
static void launch_gemm_big(hipStream_t stream,
    const bf16_t* A, int lda, long long aBS,
    const bf16_t* B, int ldb, long long bBS, int bTrans,
    const float* bias, const float* resid, float* outF, bf16_t* outB,
    int ldc, int cZdiv, long long cOuter, long long cInner,
    int M, int N, int K, int nz, int doGelu) {
  dim3 grid(N/128, M/128, nz);
  wmma_gemm_kernel<2,4,4,2><<<grid, dim3(256), 0, stream>>>(
      A, lda, aBS, B, ldb, bBS, bTrans, bias, resid, outF, outB,
      ldc, cZdiv, cOuter, cInner, K, doGelu);
}

static void launch_gemm_att(hipStream_t stream,
    const bf16_t* A, int lda, long long aBS,
    const bf16_t* B, int ldb, long long bBS, int bTrans,
    const float* bias, const float* resid, float* outF, bf16_t* outB,
    int ldc, int cZdiv, long long cOuter, long long cInner,
    int M, int N, int K, int nz, int doGelu) {
  dim3 grid(N/64, M/128, nz);
  wmma_gemm_kernel<4,1,2,4><<<grid, dim3(128), 0, stream>>>(
      A, lda, aBS, B, ldb, bBS, bTrans, bias, resid, outF, outB,
      ldc, cZdiv, cOuter, cInner, K, doGelu);
}

static void launch_cvt(hipStream_t stream, const float* src, bf16_t* dst, long long n) {
  long long blocks = (n/4 + 255) / 256;
  if (blocks > 4096) blocks = 4096;
  cvt_kernel<<<dim3((unsigned)blocks), dim3(256), 0, stream>>>(src, dst, n);
}

extern "C" void kernel_launch(void* const* d_in, const int* in_sizes, int n_in,
                              void* d_out, int out_size, void* d_ws, size_t ws_size,
                              hipStream_t stream) {
  (void)in_sizes; (void)n_in; (void)out_size; (void)ws_size;

  const float* segments  = (const float*)d_in[0];
  const float* durations = (const float*)d_in[1];
  const unsigned char* pmask = (const unsigned char*)d_in[2];
  const float* Wproj = (const float*)d_in[3];
  const float* bproj = (const float*)d_in[4];
  const float* Wdur  = (const float*)d_in[5];
  const float* bdur  = (const float*)d_in[6];
  const float* ln1g  = (const float*)d_in[7];
  const float* ln1b  = (const float*)d_in[8];
  const float* Wq    = (const float*)d_in[9];
  const float* bq    = (const float*)d_in[10];
  const float* Wk    = (const float*)d_in[11];
  const float* bk    = (const float*)d_in[12];
  const float* Wv    = (const float*)d_in[13];
  const float* bv    = (const float*)d_in[14];
  const float* Wo    = (const float*)d_in[15];
  const float* bo    = (const float*)d_in[16];
  const float* ln2g  = (const float*)d_in[17];
  const float* ln2b  = (const float*)d_in[18];
  const float* Wff1  = (const float*)d_in[19];
  const float* bff1  = (const float*)d_in[20];
  const float* Wff2  = (const float*)d_in[21];
  const float* bff2  = (const float*)d_in[22];

  // ---- workspace carve-up -------------------------------------------------
  char* ws = (char*)d_ws;
  size_t off = 0;
  auto carve = [&](size_t bytes) -> void* {
    void* p = ws + off;
    off = (off + bytes + 255) & ~(size_t)255;
    return p;
  };
  float*  xf32   = (float*) carve((size_t)MROWS*DM*4);       // residual stream
  bf16_t* seg_bf = (bf16_t*)carve((size_t)MROWS*DINC*2);
  bf16_t* h_bf   = (bf16_t*)carve((size_t)MROWS*DM*2);       // LN output (reused)
  bf16_t* wbuf   = (bf16_t*)carve((size_t)DM*FF*2);          // weight cvt buffer
  float*  qf     = (float*) carve((size_t)MROWS*DM*4);
  float*  kf     = (float*) carve((size_t)MROWS*DM*4);
  float*  vf     = (float*) carve((size_t)MROWS*DM*4);
  bf16_t* qh     = (bf16_t*)carve((size_t)BQ*NH*SQ*HDIM*2);  // [b][h][s][hd]
  bf16_t* kh     = (bf16_t*)carve((size_t)BQ*NH*SQ*HDIM*2);
  bf16_t* vh     = (bf16_t*)carve((size_t)BQ*NH*SQ*HDIM*2);
  float*  scores = (float*) carve((size_t)BQ*NH*SQ*SQ*4);    // 64 MB
  bf16_t* probs  = (bf16_t*)carve((size_t)BQ*NH*SQ*SQ*2);    // 32 MB
  bf16_t* attnb  = (bf16_t*)carve((size_t)MROWS*DM*2);
  bf16_t* ff_bf  = (bf16_t*)carve((size_t)MROWS*FF*2);

  const int ew_blocks = (MROWS*DM + 255) / 256;
  const float att_scale = 0.125f;  // 1/sqrt(64)

  // ---- embed: x = segments@Wproj + bproj + dur*Wdur + bdur ---------------
  launch_cvt(stream, segments, seg_bf, (long long)MROWS*DINC);
  launch_cvt(stream, Wproj, wbuf, (long long)DINC*DM);
  launch_gemm_big(stream, seg_bf, DINC, 0, wbuf, DM, 0, 0,
                  bproj, nullptr, xf32, nullptr, DM, 1, 0, 0,
                  MROWS, DM, DINC, 1, 0);
  embed_extras_kernel<<<dim3(ew_blocks), dim3(256), 0, stream>>>(
      xf32, durations, Wdur, bdur);

  // ---- 12 encoder layers --------------------------------------------------
  for (int l = 0; l < NL; ++l) {
    const long long wdd = (long long)l * DM * DM;
    // LN1
    ln_kernel<<<dim3(MROWS), dim3(256), 0, stream>>>(xf32, ln1g + l*DM, ln1b + l*DM, h_bf);
    // Q, K, V projections (bf16 WMMA, f32 out + bias)
    launch_cvt(stream, Wq + wdd, wbuf, (long long)DM*DM);
    launch_gemm_big(stream, h_bf, DM, 0, wbuf, DM, 0, 0,
                    bq + l*DM, nullptr, qf, nullptr, DM, 1, 0, 0, MROWS, DM, DM, 1, 0);
    launch_cvt(stream, Wk + wdd, wbuf, (long long)DM*DM);
    launch_gemm_big(stream, h_bf, DM, 0, wbuf, DM, 0, 0,
                    bk + l*DM, nullptr, kf, nullptr, DM, 1, 0, 0, MROWS, DM, DM, 1, 0);
    launch_cvt(stream, Wv + wdd, wbuf, (long long)DM*DM);
    launch_gemm_big(stream, h_bf, DM, 0, wbuf, DM, 0, 0,
                    bv + l*DM, nullptr, vf, nullptr, DM, 1, 0, 0, MROWS, DM, DM, 1, 0);
    // RoPE + per-head layout
    rope_kernel<<<dim3(ew_blocks), dim3(256), 0, stream>>>(qf, qh, 1);
    rope_kernel<<<dim3(ew_blocks), dim3(256), 0, stream>>>(kf, kh, 1);
    rope_kernel<<<dim3(ew_blocks), dim3(256), 0, stream>>>(vf, vh, 0);
    // scores[z] = q_h @ k_h^T  (z = b*H+h; K row-major per head == B^T)
    launch_gemm_big(stream, qh, HDIM, (long long)SQ*HDIM,
                    kh, HDIM, (long long)SQ*HDIM, 1,
                    nullptr, nullptr, scores, nullptr,
                    SQ, 1, (long long)SQ*SQ, 0, SQ, SQ, HDIM, BQ*NH, 0);
    // masked scaled softmax -> bf16 probs
    softmax_kernel<<<dim3(SQ, BQ*NH), dim3(256), 0, stream>>>(scores, pmask, probs, att_scale);
    // attn[z] = probs @ v_h, scattered to (s,b, h*HD+hd) bf16
    launch_gemm_att(stream, probs, SQ, (long long)SQ*SQ,
                    vh, HDIM, (long long)SQ*HDIM, 0,
                    nullptr, nullptr, nullptr, attnb,
                    BQ*DM, NH, (long long)DM, (long long)HDIM,
                    SQ, HDIM, SQ, BQ*NH, 0);
    // x = x + attn @ Wo + bo
    launch_cvt(stream, Wo + wdd, wbuf, (long long)DM*DM);
    launch_gemm_big(stream, attnb, DM, 0, wbuf, DM, 0, 0,
                    bo + l*DM, xf32, xf32, nullptr, DM, 1, 0, 0, MROWS, DM, DM, 1, 0);
    // LN2
    ln_kernel<<<dim3(MROWS), dim3(256), 0, stream>>>(xf32, ln2g + l*DM, ln2b + l*DM, h_bf);
    // FFN: gelu(h2 @ W1 + b1) @ W2 + b2, residual into x
    launch_cvt(stream, Wff1 + (long long)l*DM*FF, wbuf, (long long)DM*FF);
    launch_gemm_big(stream, h_bf, DM, 0, wbuf, FF, 0, 0,
                    bff1 + l*FF, nullptr, nullptr, ff_bf, FF, 1, 0, 0,
                    MROWS, FF, DM, 1, 1);
    launch_cvt(stream, Wff2 + (long long)l*FF*DM, wbuf, (long long)FF*DM);
    launch_gemm_big(stream, ff_bf, FF, 0, wbuf, DM, 0, 0,
                    bff2 + l*DM, xf32, xf32, nullptr, DM, 1, 0, 0,
                    MROWS, DM, FF, 1, 0);
  }

  // ---- output (f32, (S,B,D)) ---------------------------------------------
  hipMemcpyAsync(d_out, xf32, (size_t)MROWS*DM*sizeof(float),
                 hipMemcpyDeviceToDevice, stream);
}
